// Qwen2MoE_4432406249495
// MI455X (gfx1250) — matrix operand order repaired and checked by reading, not compiled, not run on hardware
//
#include <hip/hip_runtime.h>

// ---------------- problem constants ----------------
static constexpr int NTOK     = 2048;   // B*T
static constexpr int C        = 2048;
static constexpr int E        = 8;
static constexpr int F_MOE    = 1408;
static constexpr int F_SHARED = 5632;

typedef __attribute__((ext_vector_type(16))) __bf16 v16bf;
typedef __attribute__((ext_vector_type(8)))  float  v8f;

#define DEVFN __device__ __forceinline__

DEVFN unsigned short f2bf(float f) {            // fp32 -> bf16, round-nearest-even
    unsigned u = __float_as_uint(f);
    u += 0x7FFFu + ((u >> 16) & 1u);
    return (unsigned short)(u >> 16);
}
DEVFN float sigmoid_fast(float x) {             // v_rcp_f32, not IEEE divide
    return __builtin_amdgcn_rcpf(1.0f + __expf(-x));
}

DEVFN uint4 pack8(const float* v) {             // 8 fp32 -> 8 bf16 packed in uint4
    union { unsigned short s[8]; uint4 q; } u;
#pragma unroll
    for (int i = 0; i < 8; ++i) u.s[i] = f2bf(v[i]);
    return u.q;
}

// Async DMA: global -> LDS, 16 bytes per lane (ASYNCcnt tracked).
// lds_off = addr[31:0] of the shared-memory destination (LDS aperture offset).
DEVFN void async_copy_b128(unsigned lds_off, const void* gaddr) {
    asm volatile("global_load_async_to_lds_b128 %0, %1, off"
                 :: "v"(lds_off), "v"(gaddr) : "memory");
}
DEVFN void wait_async0() {
    asm volatile("s_wait_asynccnt 0" ::: "memory");
}

// A fragment: 16-bit A 16x32 layout (ISA 7.12.2): lane L holds row m=L&15,
// half kh=L>>4; VGPR j<4 -> K = 2j + 8*kh, j>=4 -> K = 16 + 2(j-4) + 8*kh.
DEVFN v16bf load_a_frag(const unsigned short* rowbase, int kh) {
    union { unsigned u[8]; v16bf v; } ua;
#pragma unroll
    for (int j = 0; j < 8; ++j) {
        int kb = ((j < 4) ? (2 * j) : (16 + 2 * (j - 4))) + kh * 8;
        ua.u[j] = *reinterpret_cast<const unsigned*>(rowbase + kb);
    }
    return ua.v;
}
// B fragment: 32x16 bf16, lane L holds col n=L&15, half kh=L>>4; VGPR j<4 -> K = 2j + 8*kh, j>=4 -> K = 16 + 2(j-4) + 8*kh.
DEVFN v16bf load_b_frag(const unsigned short* colbase, int kh) {
    union { unsigned u[8]; v16bf v; } ub;
#pragma unroll
    for (int j = 0; j < 8; ++j) {
        int kb = ((j < 4) ? (2 * j) : (16 + 2 * (j - 4))) + kh * 8;
        ub.u[j] = *reinterpret_cast<const unsigned*>(colbase + kb);
    }
    return ub.v;
}

// ---------------- router: logits, softmax, top-2, shared gate ----------------
__global__ void router_kernel(const float* __restrict__ x,
                              const float* __restrict__ gw,
                              const float* __restrict__ sgw,
                              float* __restrict__ gate,
                              int*   __restrict__ counts,
                              int*   __restrict__ tok,
                              float* __restrict__ wt) {
    int t = blockIdx.x * blockDim.x + threadIdx.x;
    if (t >= NTOK) return;
    const float* xr = x + (size_t)t * C;
    float acc[E]; float ag = 0.f;
#pragma unroll
    for (int e = 0; e < E; ++e) acc[e] = 0.f;
    for (int c = 0; c < C; ++c) {
        float xv = xr[c];
        ag += xv * sgw[c];
#pragma unroll
        for (int e = 0; e < E; ++e) acc[e] += xv * gw[(size_t)e * C + c];
    }
    float mx = acc[0];
#pragma unroll
    for (int e = 1; e < E; ++e) mx = fmaxf(mx, acc[e]);
    float s = 0.f, p[E];
#pragma unroll
    for (int e = 0; e < E; ++e) { p[e] = __expf(acc[e] - mx); s += p[e]; }
    float inv = __builtin_amdgcn_rcpf(s);
#pragma unroll
    for (int e = 0; e < E; ++e) p[e] *= inv;
    int i1 = 0; float p1 = p[0];
#pragma unroll
    for (int e = 1; e < E; ++e) if (p[e] > p1) { p1 = p[e]; i1 = e; }
    int i2 = -1; float p2 = -1.f;
#pragma unroll
    for (int e = 0; e < E; ++e) if (e != i1 && p[e] > p2) { p2 = p[e]; i2 = e; }
    int s1 = atomicAdd(&counts[i1], 1);
    tok[i1 * NTOK + s1] = t; wt[i1 * NTOK + s1] = p1;
    int s2 = atomicAdd(&counts[i2], 1);
    tok[i2 * NTOK + s2] = t; wt[i2 * NTOK + s2] = p2;
    gate[t] = sigmoid_fast(ag);
}

// ---------------- x fp32 -> bf16 ----------------
__global__ void cast_kernel(const float* __restrict__ x,
                            unsigned short* __restrict__ xb) {
    size_t i = ((size_t)blockIdx.x * blockDim.x + threadIdx.x) * 8;
    float4 a = *(const float4*)(x + i);
    float4 b = *(const float4*)(x + i + 4);
    float tmp[8] = { a.x, a.y, a.z, a.w, b.x, b.y, b.z, b.w };
    *(uint4*)(xb + i) = pack8(tmp);
}

// ---------------- SwiGLU up-projection: h = silu(x*W1^T) * (x*W2^T) ----------
// 128(M) x 64(N) block tile, K-chunk 32, double-buffered LDS, 1 barrier/chunk.
// 8 waves; each wave owns a 2x2 grid of 16x16 WMMA tiles -> 8 WMMA per chunk.
template <bool GATHER>
__global__ __launch_bounds__(256)
void mlp1_kernel(const unsigned short* __restrict__ Xb,   // [NTOK,C] bf16
                 const float* __restrict__ W1,            // [(E)|1, F, C] fp32
                 const float* __restrict__ W2,
                 unsigned short* __restrict__ H,          // [(E*NTOK)|NTOK, F] bf16
                 const int* __restrict__ counts,
                 const int* __restrict__ tokl,
                 int F) {
    constexpr int KP = 40;
    __shared__ unsigned short As [2][128][KP];
    __shared__ unsigned short B1s[2][64][KP];
    __shared__ unsigned short B2s[2][64][KP];

    int e  = blockIdx.z;
    int f0 = blockIdx.x * 64;
    int r0 = blockIdx.y * 128;
    int nrows = NTOK;
    const int* tl = nullptr;
    if (GATHER) {
        nrows = counts[e];
        if (r0 >= nrows) return;
        tl = tokl + e * NTOK;
    }
    const float* w1 = W1 + (size_t)e * F * C;
    const float* w2 = W2 + (size_t)e * F * C;

    int tid = threadIdx.x, lane = tid & 31, wave = tid >> 5;
    int mi0 = (wave & 3) * 2, nj0 = (wave >> 2) * 2;

    v8f acc1[2][2], acc2[2][2];
#pragma unroll
    for (int im = 0; im < 2; ++im)
#pragma unroll
        for (int jn = 0; jn < 2; ++jn)
#pragma unroll
            for (int r = 0; r < 8; ++r) { acc1[im][jn][r] = 0.f; acc2[im][jn][r] = 0.f; }

    // A loader: 128 rows x 32 bf16, 256 thr -> 16 bf16 each (2 async b128)
    int arr = tid >> 1, acg = (tid & 1) * 16;
    size_t arow;
    if (GATHER) {
        int slot = r0 + arr;
        int tk = (slot < nrows) ? tl[slot] : tl[0];
        arow = (size_t)tk * C;
    } else {
        arow = (size_t)(r0 + arr) * C;
    }
    // B loader: 64 rows x 32 fp32 -> bf16, 256 thr -> 8 fp32 each
    int brr = tid >> 2, bcg = (tid & 3) * 8;
    const float* w1r = w1 + (size_t)(f0 + brr) * C;
    const float* w2r = w2 + (size_t)(f0 + brr) * C;

    auto load_tile = [&](int kc, int buf) {
        unsigned l0 = (unsigned)(size_t)&As[buf][arr][acg];
        async_copy_b128(l0,      Xb + arow + kc + acg);
        async_copy_b128(l0 + 16, Xb + arow + kc + acg + 8);
        float4 a = *(const float4*)(w1r + kc + bcg);
        float4 b = *(const float4*)(w1r + kc + bcg + 4);
        float t1[8] = { a.x, a.y, a.z, a.w, b.x, b.y, b.z, b.w };
        *(uint4*)&B1s[buf][brr][bcg] = pack8(t1);
        a = *(const float4*)(w2r + kc + bcg);
        b = *(const float4*)(w2r + kc + bcg + 4);
        float t2[8] = { a.x, a.y, a.z, a.w, b.x, b.y, b.z, b.w };
        *(uint4*)&B2s[buf][brr][bcg] = pack8(t2);
    };

    load_tile(0, 0);
    wait_async0();
    __syncthreads();

    int kh = lane >> 4, ml = lane & 15;
    int buf = 0;
    for (int kc = 0; kc < C; kc += 32) {
        if (kc + 32 < C) load_tile(kc + 32, buf ^ 1);   // prefetch next chunk

        v16bf af[2], b1f[2], b2f[2];
#pragma unroll
        for (int im = 0; im < 2; ++im)
            af[im] = load_a_frag(&As[buf][(mi0 + im) * 16 + ml][0], kh);
#pragma unroll
        for (int jn = 0; jn < 2; ++jn) {
            b1f[jn] = load_b_frag(&B1s[buf][(nj0 + jn) * 16 + ml][0], kh);
            b2f[jn] = load_b_frag(&B2s[buf][(nj0 + jn) * 16 + ml][0], kh);
        }
#pragma unroll
        for (int im = 0; im < 2; ++im)
#pragma unroll
            for (int jn = 0; jn < 2; ++jn) {
                acc1[im][jn] = __builtin_amdgcn_wmma_f32_16x16x32_bf16(
                    false, af[im], false, b1f[jn], (short)0, acc1[im][jn], false, false);
                acc2[im][jn] = __builtin_amdgcn_wmma_f32_16x16x32_bf16(
                    false, af[im], false, b2f[jn], (short)0, acc2[im][jn], false, false);
            }

        wait_async0();        // my prefetch stores into buf^1 landed
        __syncthreads();      // everyone done reading buf / writing buf^1
        buf ^= 1;
    }

    int nl = lane & 15;
#pragma unroll
    for (int im = 0; im < 2; ++im)
#pragma unroll
        for (int jn = 0; jn < 2; ++jn) {
            int fc = f0 + (nj0 + jn) * 16 + nl;
#pragma unroll
            for (int r = 0; r < 8; ++r) {
                int slot = r0 + (mi0 + im) * 16 + kh * 8 + r;
                if (GATHER && slot >= nrows) continue;
                float h1 = acc1[im][jn][r];
                float hv = h1 * sigmoid_fast(h1) * acc2[im][jn][r];  // silu(h1)*h2
                size_t rowbase = GATHER ? ((size_t)e * NTOK + slot) : (size_t)slot;
                H[rowbase * F + fc] = f2bf(hv);
            }
        }
}

// ---------------- down-projection: out(+)= w * (h @ proj^T) ----------------
template <bool SCATTER>
__global__ __launch_bounds__(256)
void proj_kernel(const unsigned short* __restrict__ H,   // bf16 activations
                 const float* __restrict__ W,            // [(E)|1, C, F] fp32
                 float* __restrict__ out,
                 const float* __restrict__ gate,
                 const int* __restrict__ counts,
                 const int* __restrict__ tokl,
                 const float* __restrict__ wtl,
                 int F) {
    constexpr int KP = 40;
    __shared__ unsigned short As[2][128][KP];
    __shared__ unsigned short Bs[2][64][KP];

    int e  = blockIdx.z;
    int c0 = blockIdx.x * 64;
    int r0 = blockIdx.y * 128;
    int nrows = NTOK;
    const int* tl = nullptr; const float* wl = nullptr;
    if (SCATTER) {
        nrows = counts[e];
        if (r0 >= nrows) return;
        tl = tokl + e * NTOK; wl = wtl + e * NTOK;
    }
    const float* w = W + (size_t)e * C * F;

    int tid = threadIdx.x, lane = tid & 31, wave = tid >> 5;
    int mi0 = (wave & 3) * 2, nj0 = (wave >> 2) * 2;

    v8f acc[2][2];
#pragma unroll
    for (int im = 0; im < 2; ++im)
#pragma unroll
        for (int jn = 0; jn < 2; ++jn)
#pragma unroll
            for (int r = 0; r < 8; ++r) acc[im][jn][r] = 0.f;

    int arr = tid >> 1, acg = (tid & 1) * 16;
    size_t arow = SCATTER ? ((size_t)e * NTOK + (r0 + arr)) * F
                          : (size_t)(r0 + arr) * F;
    int brr = tid >> 2, bcg = (tid & 3) * 8;
    const float* wr = w + (size_t)(c0 + brr) * F;

    auto load_tile = [&](int kf, int buf) {
        unsigned l0 = (unsigned)(size_t)&As[buf][arr][acg];
        async_copy_b128(l0,      H + arow + kf + acg);
        async_copy_b128(l0 + 16, H + arow + kf + acg + 8);
        float4 a = *(const float4*)(wr + kf + bcg);
        float4 b = *(const float4*)(wr + kf + bcg + 4);
        float t1[8] = { a.x, a.y, a.z, a.w, b.x, b.y, b.z, b.w };
        *(uint4*)&Bs[buf][brr][bcg] = pack8(t1);
    };

    load_tile(0, 0);
    wait_async0();
    __syncthreads();

    int kh = lane >> 4, ml = lane & 15;
    int buf = 0;
    for (int kf = 0; kf < F; kf += 32) {
        if (kf + 32 < F) load_tile(kf + 32, buf ^ 1);

        v16bf af[2], bf_[2];
#pragma unroll
        for (int im = 0; im < 2; ++im)
            af[im] = load_a_frag(&As[buf][(mi0 + im) * 16 + ml][0], kh);
#pragma unroll
        for (int jn = 0; jn < 2; ++jn)
            bf_[jn] = load_b_frag(&Bs[buf][(nj0 + jn) * 16 + ml][0], kh);
#pragma unroll
        for (int im = 0; im < 2; ++im)
#pragma unroll
            for (int jn = 0; jn < 2; ++jn)
                acc[im][jn] = __builtin_amdgcn_wmma_f32_16x16x32_bf16(
                    false, af[im], false, bf_[jn], (short)0, acc[im][jn], false, false);

        wait_async0();
        __syncthreads();
        buf ^= 1;
    }

    int nl = lane & 15;
#pragma unroll
    for (int im = 0; im < 2; ++im)
#pragma unroll
        for (int jn = 0; jn < 2; ++jn) {
            int cc = c0 + (nj0 + jn) * 16 + nl;
#pragma unroll
            for (int r = 0; r < 8; ++r) {
                int slot = r0 + (mi0 + im) * 16 + kh * 8 + r;
                if (SCATTER) {
                    if (slot >= nrows) continue;
                    int tk = tl[slot];
                    atomicAdd(out + (size_t)tk * C + cc, wl[slot] * acc[im][jn][r]);
                } else {
                    out[(size_t)slot * C + cc] = gate[slot] * acc[im][jn][r];
                }
            }
        }
}

// ---------------- host launcher ----------------
extern "C" void kernel_launch(void* const* d_in, const int* in_sizes, int n_in,
                              void* d_out, int out_size, void* d_ws, size_t ws_size,
                              hipStream_t stream) {
    const float* x     = (const float*)d_in[0];
    const float* gw    = (const float*)d_in[1];
    const float* efc1  = (const float*)d_in[2];
    const float* efc2  = (const float*)d_in[3];
    const float* eproj = (const float*)d_in[4];
    const float* sfc1  = (const float*)d_in[5];
    const float* sfc2  = (const float*)d_in[6];
    const float* sproj = (const float*)d_in[7];
    const float* sgw   = (const float*)d_in[8];
    float* out = (float*)d_out;

    char* ws = (char*)d_ws;
    unsigned short* XB = (unsigned short*)ws; ws += (size_t)NTOK * C * 2;
    unsigned short* HS = (unsigned short*)ws; ws += (size_t)NTOK * F_SHARED * 2;
    unsigned short* HM = (unsigned short*)ws; ws += (size_t)E * NTOK * F_MOE * 2;
    float* GATE = (float*)ws;                 ws += (size_t)NTOK * 4;
    float* WTL  = (float*)ws;                 ws += (size_t)E * NTOK * 4;
    int*   TKL  = (int*)ws;                   ws += (size_t)E * NTOK * 4;
    int*   CNT  = (int*)ws;                   ws += 64;

    hipMemsetAsync(CNT, 0, E * sizeof(int), stream);
    router_kernel<<<NTOK / 256, 256, 0, stream>>>(x, gw, sgw, GATE, CNT, TKL, WTL);
    cast_kernel<<<((size_t)NTOK * C) / (256 * 8), 256, 0, stream>>>(x, XB);

    // shared expert (writes every out element -> initializes d_out)
    mlp1_kernel<false><<<dim3(F_SHARED / 64, NTOK / 128, 1), 256, 0, stream>>>(
        XB, sfc1, sfc2, HS, nullptr, nullptr, F_SHARED);
    proj_kernel<false><<<dim3(C / 64, NTOK / 128, 1), 256, 0, stream>>>(
        HS, sproj, out, GATE, nullptr, nullptr, nullptr, F_SHARED);

    // routed experts (top-2 only, gathered per-expert token lists)
    mlp1_kernel<true><<<dim3(F_MOE / 64, NTOK / 128, E), 256, 0, stream>>>(
        XB, efc1, efc2, HM, CNT, TKL, F_MOE);
    proj_kernel<true><<<dim3(C / 64, NTOK / 128, E), 256, 0, stream>>>(
        HM, eproj, out, nullptr, CNT, TKL, WTL, F_MOE);
}
